// ConvSP_46772193853617
// MI455X (gfx1250) — matrix operand order repaired and checked by reading, not compiled, not run on hardware
//
#include <hip/hip_runtime.h>
#include <hip/hip_bf16.h>
#include <math.h>

// ConvSP on MI455X (gfx1250): SPH kernel eval on VALU, both einsums on
// v_wmma_f32_16x16x32_f16 matrix cores, LDS-staged fragment layouts.
//
// Shapes: B=2, M=N=8192, K=32 neighbors, C=16, O=16, ncells=27.

typedef __attribute__((ext_vector_type(16))) _Float16 v16h;
typedef __attribute__((ext_vector_type(8)))  float    v8f;

#define BQ      8192
#define NB      2
#define KNBR    32
#define NCH     16
#define NOUT    16
#define NCELL   27
#define JPAD    448   // 27*16 = 432 padded up to 14 * 32
#define WAVES_PER_BLOCK 2
#define THREADS (WAVES_PER_BLOCK * 32)

__global__ __launch_bounds__(THREADS)
void convsp_wmma_kernel(const float* __restrict__ qlocs,
                        const float* __restrict__ locs,
                        const float* __restrict__ data,
                        const int*   __restrict__ nbrs,
                        const float* __restrict__ weight,
                        const float* __restrict__ bias,
                        float*       __restrict__ out)
{
    // LDS staging (48 KB total for 2 waves):
    __shared__ _Float16 sW  [NOUT][JPAD];                 // W[o][j=e*16+c], block-shared
    __shared__ _Float16 sKv [WAVES_PER_BLOCK][32][32];    // kv[e][k]   per wave (rows 27..31 zero)
    __shared__ _Float16 sNd [WAVES_PER_BLOCK][NCH][32];   // ndata^T[c][k] per wave
    __shared__ _Float16 sAcc[WAVES_PER_BLOCK][16][JPAD];  // T[t][j]    per wave

    const int lane = threadIdx.x & 31;
    const int wave = threadIdx.x >> 5;
    const int tile = blockIdx.x * WAVES_PER_BLOCK + wave;   // 1024 tiles of 16 queries
    const int b    = tile >> 9;                             // 512 tiles per batch
    const int m0   = (tile & 511) << 4;

    // ---- Stage 0: load conv weights into LDS as f16, j = e*16 + c ----------
    for (int i = threadIdx.x; i < NOUT * JPAD; i += THREADS) {
        const int o = i / JPAD, j = i % JPAD;
        float w = 0.0f;
        if (j < NCELL * NCH) {
            const int e = j >> 4, c = j & 15;
            w = weight[(o * NCH + c) * NCELL + e];
        }
        sW[o][j] = (_Float16)w;
    }
    __syncthreads();

    _Float16 (*kvq)[32]   = sKv[wave];
    _Float16 (*ndq)[32]   = sNd[wave];
    _Float16 (*accq)[JPAD]= sAcc[wave];

    // ---- Stage B: per query, kv eval (VALU) + acc = ndata^T x kv^T (WMMA) --
    for (int t = 0; t < 16; ++t) {
        const int m = m0 + t;
        const float qx = qlocs[((size_t)(b * BQ + m)) * 3 + 0];
        const float qy = qlocs[((size_t)(b * BQ + m)) * 3 + 1];
        const float qz = qlocs[((size_t)(b * BQ + m)) * 3 + 2];

        const int  nb    = nbrs[((size_t)(b * BQ + m)) * KNBR + lane];
        const bool valid = (nb >= 0);
        const int  nid   = valid ? nb : 0;

        const float* lp = locs + ((size_t)(b * BQ + nid)) * 3;
        const float nx = lp[0], ny = lp[1], nz = lp[2];

        const float* dp = data + ((size_t)(b * BQ + nid)) * NCH;
        #pragma unroll
        for (int c = 0; c < NCH; ++c)
            ndq[c][lane] = (_Float16)dp[c];                 // ndata^T[c][k=lane]

        #pragma unroll
        for (int e = 0; e < NCELL; ++e) {
            const float ox = (float)(e / 9      - 1) * 0.05f;
            const float oy = (float)((e / 3) % 3 - 1) * 0.05f;
            const float oz = (float)(e % 3      - 1) * 0.05f;
            const float dx = qx + ox - nx;
            const float dy = qy + oy - ny;
            const float dz = qz + oz - nz;
            const float d  = sqrtf(dx * dx + dy * dy + dz * dz + 1e-12f);
            const float q  = d * 10.0f;                     // d / RADIUS
            const float om = 1.0f - q;
            const float kv = (q < 1.0f && valid) ? om * om * om : 0.0f;
            kvq[e][lane] = (_Float16)kv;                    // kv[e][k=lane]
        }
        #pragma unroll
        for (int e = NCELL; e < 32; ++e)
            kvq[e][lane] = (_Float16)0.0f;                  // pad e rows

        __syncthreads();

        // A fragment: 16x32 f16, rows = channel c. Lane L<16: row c=L, K 0..7,16..23;
        // lane L>=16: row c=L-16, K 8..15,24..31.
        v16h afrag;
        {
            const int arow = lane & 15;
            const int kb   = (lane < 16) ? 0 : 8;
            #pragma unroll
            for (int h = 0; h < 16; ++h) {
                const int k = kb + (h < 8 ? h : h + 8);
                afrag[h] = ndq[arow][k];
            }
        }

        #pragma unroll
        for (int et = 0; et < 2; ++et) {
            // B fragment: 32x16 f16 = kv^T[k][e]. Lanes 0-15: K 0..7,16..23 (col e=lane);
            // lanes 16-31: K 8..15,24..31 (col e=lane-16).
            v16h bfrag;
            const int ecol = et * 16 + (lane & 15);
            const int kb2  = (lane < 16) ? 0 : 8;
            #pragma unroll
            for (int h = 0; h < 16; ++h)
                bfrag[h] = kvq[ecol][kb2 + (h < 8 ? h : h + 8)];

            v8f cc = {};
            cc = __builtin_amdgcn_wmma_f32_16x16x32_f16(
                     false, afrag, false, bfrag, (short)0, cc, false, false);

            // C layout: vgpr r, lanes 0-15 -> c=r; lanes 16-31 -> c=r+8; col e.
            const int coff = (lane < 16) ? 0 : 8;
            if (ecol < NCELL) {
                #pragma unroll
                for (int r = 0; r < 8; ++r)
                    accq[t][ecol * 16 + coff + r] = (_Float16)cc[r];
            }
        }
        if (lane < 16)
            accq[t][NCELL * NCH + lane] = (_Float16)0.0f;   // zero j-pad 432..447
        __syncthreads();
    }

    // ---- Stage C: out(16x16) = T(16x448) x W(448x16), 14 accumulating WMMAs -
    v8f oacc = {};
    #pragma unroll
    for (int s = 0; s < JPAD / 32; ++s) {
        v16h af, bf;
        const int trow = lane & 15;            // query row
        const int kb   = (lane < 16) ? 0 : 8;
        #pragma unroll
        for (int h = 0; h < 16; ++h)
            af[h] = accq[trow][s * 32 + kb + (h < 8 ? h : h + 8)];

        const int ocol = lane & 15;            // output column o
        const int kb2  = (lane < 16) ? 0 : 8;
        #pragma unroll
        for (int h = 0; h < 16; ++h)
            bf[h] = sW[ocol][s * 32 + kb2 + (h < 8 ? h : h + 8)];

        oacc = __builtin_amdgcn_wmma_f32_16x16x32_f16(
                   false, af, false, bf, (short)0, oacc, false, false);
    }

    // ---- Epilogue: bias + store. C layout: vgpr r -> query t=r(+8), col -> o.
    const int o  = lane & 15;
    const float bv = bias[o];
    const int tb = (lane < 16) ? 0 : 8;
    #pragma unroll
    for (int r = 0; r < 8; ++r) {
        const int t = tb + r;
        out[((size_t)(b * BQ + m0 + t)) * NOUT + o] = oacc[r] + bv;
    }
}

extern "C" void kernel_launch(void* const* d_in, const int* in_sizes, int n_in,
                              void* d_out, int out_size, void* d_ws, size_t ws_size,
                              hipStream_t stream) {
    (void)in_sizes; (void)n_in; (void)out_size; (void)d_ws; (void)ws_size;
    const float* qlocs  = (const float*)d_in[0];
    const float* locs   = (const float*)d_in[1];
    const float* data   = (const float*)d_in[2];
    const int*   nbrs   = (const int*)  d_in[3];
    const float* weight = (const float*)d_in[4];
    const float* bias   = (const float*)d_in[5];
    float*       out    = (float*)d_out;

    // 1024 query-tiles of 16, 2 waves (tiles) per block -> 512 blocks x 64 thr.
    const int tiles  = (NB * BQ) / 16;
    const int blocks = tiles / WAVES_PER_BLOCK;
    convsp_wmma_kernel<<<blocks, THREADS, 0, stream>>>(
        qlocs, locs, data, nbrs, weight, bias, out);
}
